// MambaMapping_14508399526511
// MI455X (gfx1250) — matrix operand order repaired and checked by reading, not compiled, not run on hardware
//
#include <hip/hip_runtime.h>
#include <hip/hip_bf16.h>

#define BSZ   64
#define TSEQ  40
#define EDIM  768
#define DIN   1536     // d_inner
#define DST   16       // d_state
#define DTRK  48       // dt_rank
#define DTRP  64       // dt_rank padded to mult of 32 for WMMA K
#define XDBCN 80       // dt_rank + 2*d_state (valid cols of x_proj output)
#define XDBCP 128      // x_proj N padded to mult of 64 -> branch-free GEMM
#define NLAY  8
#define NROWS (BSZ * TSEQ)   // 2560
#define FCN   (TSEQ * EDIM)  // 30720

typedef __attribute__((ext_vector_type(16))) _Float16 v16h;
typedef __attribute__((ext_vector_type(8)))  float    v8f;

union HFrag { v16h h; uint4 q[2]; };

__device__ __forceinline__ float softplusf(float x) {
  return (x > 20.f) ? x : log1pf(__expf(x));
}
__device__ __forceinline__ float siluf(float x) {
  return x / (1.f + __expf(-x));
}

// ---------------------------------------------------------------------------
// f16 WMMA GEMM:  C[M,N] = A[M,K] * B[N,K]^T (+bias[N]) (+softplus)
// A,B f16 row-major (K contiguous); C f32.
// Requirements: M%64==0, K%32==0, N a multiple of 64 (padded upstream).
// Wave tile: 32(M) x 64(N) -> 8 WMMAs per k-step, 2 A frags + 4 B frags.
// Block = 8 waves (2 in M x 4 in N) = 64(M) x 256(N).
// Branch-free inner loop: WMMA always runs with EXEC all-ones.
// ---------------------------------------------------------------------------
__global__ __launch_bounds__(256) void gemm_f16_wmma(
    const _Float16* __restrict__ A,
    const _Float16* __restrict__ B,
    const float* __restrict__ bias,   // nullptr allowed
    float* __restrict__ C,
    int M, int N, int K, int act)     // act: 0 none, 1 softplus
{
  const int lane = threadIdx.x & 31;
  const int wave = threadIdx.x >> 5;
  const int wm = wave >> 2;           // 0..1
  const int wn = wave & 3;            // 0..3
  const int m0 = blockIdx.x * 64 + wm * 32;
  const int n0 = blockIdx.y * 256 + wn * 64;
  if (m0 >= M || n0 >= N) return;     // wave-uniform: EXEC all-0 or all-1

  const int r  = lane & 15;
  const int hi = lane >> 4;

  v8f acc[2][4] = {};
  const _Float16* Arow0 = A + (size_t)(m0 + r) * K + (hi << 3);
  const _Float16* Arow1 = Arow0 + (size_t)16 * K;
  const _Float16* Bp[4];
#pragma unroll
  for (int t = 0; t < 4; ++t)
    Bp[t] = B + (size_t)(n0 + t * 16 + r) * K + (hi << 3);

#pragma unroll 2
  for (int k0 = 0; k0 < K; k0 += 32) {
    // Issue all loads of this k-step, then 8 WMMAs.
    HFrag a0, a1, b0, b1, b2, b3;
    a0.q[0] = *(const uint4*)(Arow0 + k0);
    a0.q[1] = *(const uint4*)(Arow0 + k0 + 16);
    a1.q[0] = *(const uint4*)(Arow1 + k0);
    a1.q[1] = *(const uint4*)(Arow1 + k0 + 16);
    b0.q[0] = *(const uint4*)(Bp[0] + k0);
    b0.q[1] = *(const uint4*)(Bp[0] + k0 + 16);
    b1.q[0] = *(const uint4*)(Bp[1] + k0);
    b1.q[1] = *(const uint4*)(Bp[1] + k0 + 16);
    b2.q[0] = *(const uint4*)(Bp[2] + k0);
    b2.q[1] = *(const uint4*)(Bp[2] + k0 + 16);
    b3.q[0] = *(const uint4*)(Bp[3] + k0);
    b3.q[1] = *(const uint4*)(Bp[3] + k0 + 16);
    __builtin_prefetch(Arow0 + k0 + 32, 0, 1);
    acc[0][0] = __builtin_amdgcn_wmma_f32_16x16x32_f16(
        false, a0.h, false, b0.h, (short)0, acc[0][0], false, false);
    acc[1][0] = __builtin_amdgcn_wmma_f32_16x16x32_f16(
        false, a1.h, false, b0.h, (short)0, acc[1][0], false, false);
    acc[0][1] = __builtin_amdgcn_wmma_f32_16x16x32_f16(
        false, a0.h, false, b1.h, (short)0, acc[0][1], false, false);
    acc[1][1] = __builtin_amdgcn_wmma_f32_16x16x32_f16(
        false, a1.h, false, b1.h, (short)0, acc[1][1], false, false);
    acc[0][2] = __builtin_amdgcn_wmma_f32_16x16x32_f16(
        false, a0.h, false, b2.h, (short)0, acc[0][2], false, false);
    acc[1][2] = __builtin_amdgcn_wmma_f32_16x16x32_f16(
        false, a1.h, false, b2.h, (short)0, acc[1][2], false, false);
    acc[0][3] = __builtin_amdgcn_wmma_f32_16x16x32_f16(
        false, a0.h, false, b3.h, (short)0, acc[0][3], false, false);
    acc[1][3] = __builtin_amdgcn_wmma_f32_16x16x32_f16(
        false, a1.h, false, b3.h, (short)0, acc[1][3], false, false);
  }

#pragma unroll
  for (int u = 0; u < 2; ++u) {
#pragma unroll
    for (int t = 0; t < 4; ++t) {
      const int n = n0 + t * 16 + r;
      const float bv = bias ? bias[n] : 0.f;
#pragma unroll
      for (int v = 0; v < 8; ++v) {
        const int m = m0 + u * 16 + v + (hi << 3);
        float x = acc[u][t][v] + bv;
        if (act == 1) x = softplusf(x);
        C[(size_t)m * N + n] = x;
      }
    }
  }
}

// ---------------------------------------------------------------------------
// Elementwise / reduction kernels
// ---------------------------------------------------------------------------
__global__ void k_f32_to_f16(const float* __restrict__ in,
                             _Float16* __restrict__ out, int n) {
  int i = blockIdx.x * 256 + threadIdx.x;
  if (i < n) out[i] = (_Float16)in[i];
}

// generic row-padded f32->f16 weight convert: in [rows_in,K] -> out [rows_out,K]
__global__ void k_cvt_pad_rows(const float* __restrict__ in,
                               _Float16* __restrict__ out,
                               int rows_in, int rows_out, int K) {
  int i = blockIdx.x * 256 + threadIdx.x;
  if (i >= rows_out * K) return;
  int row = i / K;
  out[i] = (row < rows_in) ? (_Float16)in[i] : (_Float16)0.f;
}

// dt_proj_w [DIN,48] -> f16 [DIN,64] zero-padded columns
__global__ void k_cvt_pad_dtw(const float* __restrict__ w,
                              _Float16* __restrict__ out) {
  int i = blockIdx.x * 256 + threadIdx.x;
  if (i >= DIN * DTRP) return;
  int row = i >> 6, col = i & 63;
  out[i] = (col < DTRK) ? (_Float16)w[row * DTRK + col] : (_Float16)0.f;
}

// xdbc[:, :48] -> f16 [NROWS,64] zero-padded columns
__global__ void k_make_dtpad(const float* __restrict__ xdbc,
                             _Float16* __restrict__ out) {
  int i = blockIdx.x * 256 + threadIdx.x;
  if (i >= NROWS * DTRP) return;
  int row = i >> 6, col = i & 63;
  out[i] = (col < DTRK) ? (_Float16)xdbc[row * XDBCP + col] : (_Float16)0.f;
}

// BatchNorm stats per prefix position p over (batch, embed)
__global__ __launch_bounds__(256) void k_bn_stats(
    const float* __restrict__ h, float* __restrict__ mu,
    float* __restrict__ rstd) {
  const int p = blockIdx.x;
  float s = 0.f, s2 = 0.f;
  for (int i = threadIdx.x; i < BSZ * EDIM; i += 256) {
    int b = i / EDIM, e = i - b * EDIM;
    float v = h[((size_t)b * TSEQ + p) * EDIM + e];
    s += v; s2 += v * v;
  }
  __shared__ float sh1[256], sh2[256];
  sh1[threadIdx.x] = s; sh2[threadIdx.x] = s2;
  __syncthreads();
  for (int off = 128; off > 0; off >>= 1) {
    if (threadIdx.x < off) {
      sh1[threadIdx.x] += sh1[threadIdx.x + off];
      sh2[threadIdx.x] += sh2[threadIdx.x + off];
    }
    __syncthreads();
  }
  if (threadIdx.x == 0) {
    const float inv = 1.f / (float)(BSZ * EDIM);
    float m = sh1[0] * inv;
    float var = sh2[0] * inv - m * m;
    mu[p] = m;
    rstd[p] = rsqrtf(var + 1e-5f);
  }
}

// h = silu((h-mu)*rstd) * mask  (in place), plus f16 copy for next GEMM
__global__ void k_bn_apply(float* __restrict__ h, const float* __restrict__ mu,
                           const float* __restrict__ rstd,
                           const float* __restrict__ mask,
                           _Float16* __restrict__ x16, int n) {
  int i = blockIdx.x * 256 + threadIdx.x;
  if (i >= n) return;
  int p = (i / EDIM) % TSEQ;
  float v = (h[i] - mu[p]) * rstd[p];
  v = siluf(v);
  v *= mask[i];
  h[i] = v;
  x16[i] = (_Float16)v;
}

// causal depthwise conv1d (DC=4) + bias + SiLU over xz[:, 0:DIN]
__global__ __launch_bounds__(256) void k_conv_silu(
    const float* __restrict__ xz, const float* __restrict__ cw,
    const float* __restrict__ cb, float* __restrict__ xi,
    _Float16* __restrict__ xi16) {
  int idx = blockIdx.x * 256 + threadIdx.x;
  if (idx >= BSZ * DIN) return;
  int b = idx / DIN, d = idx - b * DIN;
  const float w0 = cw[d * 4 + 0], w1 = cw[d * 4 + 1];
  const float w2 = cw[d * 4 + 2], w3 = cw[d * 4 + 3];
  const float bias = cb[d];
  float x0 = 0.f, x1 = 0.f, x2 = 0.f;
  const float* base = xz + (size_t)b * TSEQ * (2 * DIN) + d;
  for (int t = 0; t < TSEQ; ++t) {
    float x3 = base[(size_t)t * (2 * DIN)];
    float v = siluf(w0 * x0 + w1 * x1 + w2 * x2 + w3 * x3 + bias);
    size_t o = ((size_t)b * TSEQ + t) * DIN + d;
    xi[o] = v;
    xi16[o] = (_Float16)v;
    x0 = x1; x1 = x2; x2 = x3;
  }
}

// selective scan: thread = (b, d); 16 states in registers; fused D-skip +
// SiLU(z) gate; emits f16 for out_proj GEMM.
__global__ __launch_bounds__(256) void k_scan(
    const float* __restrict__ dt, const float* __restrict__ xi,
    const float* __restrict__ xdbc, const float* __restrict__ xz,
    const float* __restrict__ A_log, const float* __restrict__ Dp,
    _Float16* __restrict__ y16) {
  int idx = blockIdx.x * 256 + threadIdx.x;
  if (idx >= BSZ * DIN) return;
  int b = idx / DIN, d = idx - b * DIN;
  float Ac[DST];
#pragma unroll
  for (int n = 0; n < DST; ++n) Ac[n] = -__expf(A_log[d * DST + n]);
  const float Dv = Dp[d];
  float h[DST];
#pragma unroll
  for (int n = 0; n < DST; ++n) h[n] = 0.f;
  for (int t = 0; t < TSEQ; ++t) {
    const size_t row = (size_t)b * TSEQ + t;
    const float dtv = dt[row * DIN + d];
    const float xiv = xi[row * DIN + d];
    const float zv  = xz[row * (2 * DIN) + DIN + d];
    const float* bc = xdbc + row * XDBCP;
    const float dbx = dtv * xiv;
    float y = 0.f;
#pragma unroll
    for (int n = 0; n < DST; ++n) {
      float dA = __expf(dtv * Ac[n]);
      h[n] = dA * h[n] + dbx * bc[DTRK + n];        // B
      y += h[n] * bc[DTRK + DST + n];               // C
    }
    y += xiv * Dv;
    y *= siluf(zv);
    y16[row * DIN + d] = (_Float16)y;
  }
}

__global__ __launch_bounds__(256) void k_rmsnorm(
    const float* __restrict__ x, const float* __restrict__ w,
    float* __restrict__ out) {
  const int row = blockIdx.x;
  __shared__ float sh[256];
  const float* xr = x + (size_t)row * EDIM;
  float s = 0.f;
  for (int e = threadIdx.x; e < EDIM; e += 256) { float v = xr[e]; s += v * v; }
  sh[threadIdx.x] = s;
  __syncthreads();
  for (int off = 128; off > 0; off >>= 1) {
    if (threadIdx.x < off) sh[threadIdx.x] += sh[threadIdx.x + off];
    __syncthreads();
  }
  const float rms = rsqrtf(sh[0] / (float)EDIM + 1e-5f);
  for (int e = threadIdx.x; e < EDIM; e += 256)
    out[(size_t)row * EDIM + e] = xr[e] * rms * w[e];
}

// ---------------------------------------------------------------------------
extern "C" void kernel_launch(void* const* d_in, const int* in_sizes, int n_in,
                              void* d_out, int out_size, void* d_ws,
                              size_t ws_size, hipStream_t stream) {
  const float* src   = (const float*)d_in[0];
  const float* fc_w  = (const float*)d_in[1];
  const float* fc_b  = (const float*)d_in[2];
  const float* dmask = (const float*)d_in[3];
  const float* inW   = (const float*)d_in[4];
  const float* convW = (const float*)d_in[5];
  const float* convB = (const float*)d_in[6];
  const float* xpW   = (const float*)d_in[7];
  const float* dtW   = (const float*)d_in[8];
  const float* dtB   = (const float*)d_in[9];
  const float* Alog  = (const float*)d_in[10];
  const float* Dpar  = (const float*)d_in[11];
  const float* outW  = (const float*)d_in[12];
  const float* normW = (const float*)d_in[13];

  char* ws = (char*)d_ws;
  size_t off = 0;
  auto carve = [&](size_t bytes) -> void* {
    void* p = ws + off;
    off += (bytes + 255) & ~(size_t)255;
    return p;
  };

  _Float16* fcW16  = (_Float16*)carve((size_t)FCN * EDIM * 2);
  _Float16* src16  = (_Float16*)carve((size_t)BSZ * EDIM * 2);
  _Float16* inW16  = (_Float16*)carve((size_t)2 * DIN * EDIM * 2);
  _Float16* xpW16  = (_Float16*)carve((size_t)XDBCP * DIN * 2);
  _Float16* dtW16  = (_Float16*)carve((size_t)DIN * DTRP * 2);
  _Float16* outW16 = (_Float16*)carve((size_t)EDIM * DIN * 2);
  float*    x      = (float*)   carve((size_t)NROWS * EDIM * 4);
  float*    mu     = (float*)   carve((size_t)TSEQ * 4);
  float*    rstd   = (float*)   carve((size_t)TSEQ * 4);
  _Float16* x16    = (_Float16*)carve((size_t)NROWS * EDIM * 2);
  float*    xz     = (float*)   carve((size_t)NROWS * 2 * DIN * 4);
  float*    xi     = (float*)   carve((size_t)NROWS * DIN * 4);
  _Float16* xi16   = (_Float16*)carve((size_t)NROWS * DIN * 2);
  float*    xdbc   = (float*)   carve((size_t)NROWS * XDBCP * 4);
  _Float16* dtpad  = (_Float16*)carve((size_t)NROWS * DTRP * 2);
  float*    dtbuf  = (float*)   carve((size_t)NROWS * DIN * 4);
  _Float16* y16    = (_Float16*)carve((size_t)NROWS * DIN * 2);

  auto cvt = [&](const float* in, _Float16* out, int n) {
    k_f32_to_f16<<<dim3((n + 255) / 256), dim3(256), 0, stream>>>(in, out, n);
  };
  auto gemm = [&](const _Float16* A, const _Float16* B, const float* bias,
                  float* C, int M, int N, int K, int act) {
    gemm_f16_wmma<<<dim3(M / 64, (N + 255) / 256), dim3(256), 0, stream>>>(
        A, B, bias, C, M, N, K, act);
  };

  // --- prefix: fc GEMM + BatchNorm + SiLU + dropout ---
  cvt(fc_w, fcW16, FCN * EDIM);
  cvt(src, src16, BSZ * EDIM);
  gemm(src16, fcW16, fc_b, x, BSZ, FCN, EDIM, 0);
  k_bn_stats<<<dim3(TSEQ), dim3(256), 0, stream>>>(x, mu, rstd);
  k_bn_apply<<<dim3((NROWS * EDIM + 255) / 256), dim3(256), 0, stream>>>(
      x, mu, rstd, dmask, x16, NROWS * EDIM);

  // --- 8 Mamba layers ---
  for (int l = 0; l < NLAY; ++l) {
    cvt(inW + (size_t)l * 2 * DIN * EDIM, inW16, 2 * DIN * EDIM);
    k_cvt_pad_rows<<<dim3((XDBCP * DIN + 255) / 256), dim3(256), 0, stream>>>(
        xpW + (size_t)l * XDBCN * DIN, xpW16, XDBCN, XDBCP, DIN);
    cvt(outW + (size_t)l * EDIM * DIN, outW16, EDIM * DIN);
    k_cvt_pad_dtw<<<dim3((DIN * DTRP + 255) / 256), dim3(256), 0, stream>>>(
        dtW + (size_t)l * DIN * DTRK, dtW16);

    // in_proj
    gemm(x16, inW16, nullptr, xz, NROWS, 2 * DIN, EDIM, 0);
    // causal depthwise conv + SiLU
    k_conv_silu<<<dim3((BSZ * DIN + 255) / 256), dim3(256), 0, stream>>>(
        xz, convW + (size_t)l * DIN * 4, convB + (size_t)l * DIN, xi, xi16);
    // x_proj -> [dt | B | C | zero-pad] (N padded 80 -> 128)
    gemm(xi16, xpW16, nullptr, xdbc, NROWS, XDBCP, DIN, 0);
    // dt_proj (K padded 48->64) + bias + softplus
    k_make_dtpad<<<dim3((NROWS * DTRP + 255) / 256), dim3(256), 0, stream>>>(
        xdbc, dtpad);
    gemm(dtpad, dtW16, dtB + (size_t)l * DIN, dtbuf, NROWS, DIN, DTRP, 1);
    // selective scan + D-skip + SiLU(z) gate
    k_scan<<<dim3((BSZ * DIN + 255) / 256), dim3(256), 0, stream>>>(
        dtbuf, xi, xdbc, xz, Alog + (size_t)l * DIN * DST,
        Dpar + (size_t)l * DIN, y16);
    // out_proj -> next x
    gemm(y16, outW16, nullptr, x, NROWS, EDIM, DIN, 0);
    cvt(x, x16, NROWS * EDIM);
  }

  // --- RMSNorm ---
  k_rmsnorm<<<dim3(NROWS), dim3(256), 0, stream>>>(x, normW, (float*)d_out);
}
